// GCN_39960375722765
// MI455X (gfx1250) — matrix operand order repaired and checked by reading, not compiled, not run on hardware
//
#include <hip/hip_runtime.h>
#include <hip/hip_bf16.h>

typedef __bf16 bf16_t;
typedef __attribute__((ext_vector_type(8)))  __bf16 v8bf;
typedef __attribute__((ext_vector_type(16))) __bf16 v16bf;
typedef __attribute__((ext_vector_type(8)))  float  v8f;

constexpr int  NNODE = 50000;
constexpr int  NFEAT = 256;
constexpr int  NHID  = 128;
constexpr int  NCLS  = 40;
constexpr int  NCLSP = 48;      // class dim padded to 3 WMMA tiles
constexpr int  NEDGE = 800000;

// ---- workspace layout (bytes) ----
constexpr size_t OFF_A   = 0;            // xbf16 (25.6MB) -> q f32 -> logits f32
constexpr size_t OFF_B   = 25600000;     // support1 f32 -> k f32 -> support2 f32
constexpr size_t OFF_C   = 51200000;     // h1 f32 (GCN1 agg, relu'd; CRF anchor)
constexpr size_t OFF_D   = 76800000;     // h bf16 (h1 then h2)
constexpr size_t OFF_E   = 89600000;     // hcur f32 (CRF state / final h2)
constexpr size_t OFF_F   = 115200000;    // CRF agg f32
constexpr size_t OFF_G   = 140800000;    // edge scores / exp values
constexpr size_t OFF_H   = 144000000;    // per-node max (mapped uint)
constexpr size_t OFF_I   = 144200000;    // per-node softmax denom z
constexpr size_t OFF_W1T = 144400000;    // W1^T bf16 [128][256]
constexpr size_t OFF_WQT = OFF_W1T + 65536;
constexpr size_t OFF_WKT = OFF_WQT + 32768;
constexpr size_t OFF_W2T = OFF_WKT + 32768;   // [48][128], rows >=40 zero
constexpr size_t OFF_B2P = OFF_W2T + 12288;   // padded bias2 [48]

// ------------------- small prep kernels -------------------
__global__ void k_f2bf(const float* __restrict__ in, bf16_t* __restrict__ out, long n) {
  long i = (long)blockIdx.x * blockDim.x + threadIdx.x;
  if (i < n) out[i] = (bf16_t)in[i];
}

// Wt[n*K + k] = W[k*Nin + n] (bf16), zero-padded for n >= Nin
__global__ void k_wt_bf16(const float* __restrict__ W, bf16_t* __restrict__ Wt,
                          int K, int Nin, int Npad) {
  long i = (long)blockIdx.x * blockDim.x + threadIdx.x;
  long total = (long)K * Npad;
  if (i >= total) return;
  int n = (int)(i % Npad);
  int k = (int)(i / Npad);
  float v = (n < Nin) ? W[(long)k * Nin + n] : 0.0f;
  Wt[(long)n * K + k] = (bf16_t)v;
}

__global__ void k_padb(const float* __restrict__ b, float* __restrict__ bp, int n, int npad) {
  int i = blockIdx.x * blockDim.x + threadIdx.x;
  if (i < npad) bp[i] = (i < n) ? b[i] : 0.0f;
}

// ------------------- WMMA GEMM (bf16 -> f32) -------------------
// C[M,Nout] = A[M,K] @ Bt[Nout,K]^T + bias ; one wave per 16x16 tile.
// A row-major bf16, Bt is K-major (pre-transposed) bf16. M%16==0, K%32==0.
// Block = 8 waves sharing one 16-row A tile, staged to LDS with the CDNA5
// async-to-LDS path (ASYNCcnt), then consumed via ds_load_b128.
__global__ void __launch_bounds__(256)
k_wmma_gemm(const bf16_t* __restrict__ A, const bf16_t* __restrict__ Bt,
            const float* __restrict__ bias, float* __restrict__ C,
            int M, int K, int Nout) {
  __shared__ bf16_t As[16 * 256];   // up to K=256 -> 8KB per block
  const int tid  = threadIdx.x;
  const int wave = tid >> 5;
  const int lane = tid & 31;
  const int m0 = blockIdx.x * 16;

  // --- stage A tile (16 x K bf16) into LDS: 16B per lane-transfer, async ---
  {
    const int nvec = (16 * K) / 8;                 // # of 8-bf16 (16B) chunks
    for (int i = tid; i < nvec; i += 256) {
      const bf16_t* g = A + (long)m0 * K + i * 8;  // contiguous rows in tile
      unsigned lofs = (unsigned)(size_t)(&As[i * 8]);  // low 32 bits = LDS offset
      asm volatile("global_load_async_to_lds_b128 %0, %1, off"
                   :: "v"(lofs), "v"(g) : "memory");
    }
    asm volatile("s_wait_asynccnt 0x0" ::: "memory");
  }
  __syncthreads();   // all waves (incl. inactive n-tiles) reach this

  const int n0 = (blockIdx.y * 8 + wave) * 16;
  if (n0 >= Nout) return;

  const int laneHi = lane >> 4;       // 0 or 1
  const int l15 = lane & 15;
  const bf16_t* arow = As + l15 * K;             // LDS, row of this lane
  const long brow = (long)(n0 + l15) * K;
  const int kbA = laneHi * 8;         // ISA 16-bit A 16x32 layout
  const int kbB = laneHi * 8;         // same k order as A

  v8f c = {};
  for (int k0 = 0; k0 < K; k0 += 32) {
    union { v16bf v; v8bf h[2]; } a;
    a.h[0] = *(const v8bf*)(arow + k0 + kbA);            // ds_load_b128
    a.h[1] = *(const v8bf*)(arow + k0 + kbA + 16);       // ds_load_b128
    union { v16bf v; v8bf h[2]; } bu;
    bu.h[0] = *(const v8bf*)(Bt + brow + k0 + kbB);
    bu.h[1] = *(const v8bf*)(Bt + brow + k0 + kbB + 16);
    v16bf b = bu.v;                                      // global (L2-resident)
    c = __builtin_amdgcn_wmma_f32_16x16x32_bf16(
        /*neg_a=*/false, a.v, /*neg_b=*/false, b,
        /*c_mod=*/(short)0, c, /*reuse_a=*/false, /*reuse_b=*/false);
  }
  float bb = bias ? bias[n0 + l15] : 0.0f;
  const long cbase = (long)(m0 + laneHi * 8) * Nout + n0 + l15;
#pragma unroll
  for (int i = 0; i < 8; i++) C[cbase + (long)i * Nout] = c[i] + bb;   // row m0+laneHi*8+i
}

// ------------------- edge kernels -------------------
__global__ void k_relu(float* __restrict__ p, long n) {
  long i = (long)blockIdx.x * blockDim.x + threadIdx.x;
  if (i < n) p[i] = fmaxf(p[i], 0.0f);
}

// out[dst] += w_e * sup[src], 128 features, one wave per edge
__global__ void __launch_bounds__(256)
k_spmm128(const float* __restrict__ sup, const int* __restrict__ src,
          const int* __restrict__ dst, const float* __restrict__ ew,
          float* __restrict__ out) {
  int e = blockIdx.x * 8 + (threadIdx.x >> 5);
  int lane = threadIdx.x & 31;
  if (e >= NEDGE) return;
  int s = src[e], d = dst[e];
  float w = ew[e];
  float4 v = ((const float4*)(sup + (long)s * NHID))[lane];
  float* o = out + (long)d * NHID + lane * 4;
  atomicAdd(o + 0, w * v.x);
  atomicAdd(o + 1, w * v.y);
  atomicAdd(o + 2, w * v.z);
  atomicAdd(o + 3, w * v.w);
}

__device__ inline unsigned map_f(float f) {
  unsigned u = __float_as_uint(f);
  return (u & 0x80000000u) ? ~u : (u | 0x80000000u);
}
__device__ inline float unmap_f(unsigned u) {
  return __uint_as_float((u & 0x80000000u) ? (u ^ 0x80000000u) : ~u);
}

// scores[e] = dot(q[dst], k[src]) / sqrt(128); atomicMax per dst
__global__ void __launch_bounds__(256)
k_edge_scores(const float* __restrict__ q, const float* __restrict__ kk,
              const int* __restrict__ src, const int* __restrict__ dst,
              float* __restrict__ scores, unsigned* __restrict__ maxu) {
  int e = blockIdx.x * 8 + (threadIdx.x >> 5);
  int lane = threadIdx.x & 31;
  if (e >= NEDGE) return;
  int s = src[e], d = dst[e];
  float4 a = ((const float4*)(q + (long)d * NHID))[lane];
  float4 b = ((const float4*)(kk + (long)s * NHID))[lane];
  float acc = a.x * b.x + a.y * b.y + a.z * b.z + a.w * b.w;
  for (int off = 16; off; off >>= 1) acc += __shfl_xor(acc, off, 32);
  if (lane == 0) {
    float sc = acc * 0.08838834764831843f;  // 1/sqrt(128)
    scores[e] = sc;
    atomicMax(maxu + d, map_f(sc));
  }
}

// scores[e] = exp(scores[e] - m[dst]); z[dst] += scores[e]
__global__ void k_edge_exp(float* __restrict__ scores, const int* __restrict__ dst,
                           const unsigned* __restrict__ maxu, float* __restrict__ z) {
  int e = blockIdx.x * blockDim.x + threadIdx.x;
  if (e >= NEDGE) return;
  int d = dst[e];
  float ev = expf(scores[e] - unmap_f(maxu[d]));
  scores[e] = ev;
  atomicAdd(z + d, ev);
}

// agg[dst] += (e_e / (z[dst]+1e-16)) * h[src], 128 features
__global__ void __launch_bounds__(256)
k_crf_agg(const float* __restrict__ h, const float* __restrict__ scores,
          const float* __restrict__ z, const int* __restrict__ src,
          const int* __restrict__ dst, float* __restrict__ agg) {
  int e = blockIdx.x * 8 + (threadIdx.x >> 5);
  int lane = threadIdx.x & 31;
  if (e >= NEDGE) return;
  int s = src[e], d = dst[e];
  float g = scores[e] / (z[d] + 1e-16f);
  float4 v = ((const float4*)(h + (long)s * NHID))[lane];
  float* o = agg + (long)d * NHID + lane * 4;
  atomicAdd(o + 0, g * v.x);
  atomicAdd(o + 1, g * v.y);
  atomicAdd(o + 2, g * v.z);
  atomicAdd(o + 3, g * v.w);
}

__global__ void k_crf_update(const float* __restrict__ hanchor, const float* __restrict__ agg,
                             const float* __restrict__ alpha, const float* __restrict__ beta,
                             float* __restrict__ hout, long n) {
  long i = (long)blockIdx.x * blockDim.x + threadIdx.x;
  if (i >= n) return;
  float a = alpha[0], b = beta[0];
  hout[i] = (a * hanchor[i] + b * agg[i]) / (a + b);
}

// out[dst] += w_e * sup2[src] over 40 classes (sup2 padded ld=48)
__global__ void __launch_bounds__(256)
k_spmm40(const float* __restrict__ sup2, const int* __restrict__ src,
         const int* __restrict__ dst, const float* __restrict__ ew,
         float* __restrict__ out) {
  int e = blockIdx.x * 8 + (threadIdx.x >> 5);
  int lane = threadIdx.x & 31;
  if (e >= NEDGE) return;
  int s = src[e], d = dst[e];
  float w = ew[e];
  for (int f = lane; f < NCLS; f += 32)
    atomicAdd(out + (long)d * NCLS + f, w * sup2[(long)s * NCLSP + f]);
}

// one wave per node; 40 classes
__global__ void __launch_bounds__(256)
k_log_softmax(const float* __restrict__ logits, float* __restrict__ out) {
  int n = blockIdx.x * 8 + (threadIdx.x >> 5);
  int lane = threadIdx.x & 31;
  if (n >= NNODE) return;
  const float* row = logits + (long)n * NCLS;
  float v0 = row[lane];
  float v1 = (lane < 8) ? row[32 + lane] : -3.4e38f;
  float m = fmaxf(v0, v1);
  for (int off = 16; off; off >>= 1) m = fmaxf(m, __shfl_xor(m, off, 32));
  float se = expf(v0 - m) + ((lane < 8) ? expf(v1 - m) : 0.0f);
  for (int off = 16; off; off >>= 1) se += __shfl_xor(se, off, 32);
  float ls = logf(se);
  float* orow = out + (long)n * NCLS;
  orow[lane] = v0 - m - ls;
  if (lane < 8) orow[32 + lane] = v1 - m - ls;
}

// ------------------- driver -------------------
extern "C" void kernel_launch(void* const* d_in, const int* in_sizes, int n_in,
                              void* d_out, int out_size, void* d_ws, size_t ws_size,
                              hipStream_t stream) {
  const float* x     = (const float*)d_in[0];
  const float* ew    = (const float*)d_in[1];
  const float* W1    = (const float*)d_in[2];
  const float* b1    = (const float*)d_in[3];
  const float* W2    = (const float*)d_in[4];
  const float* b2    = (const float*)d_in[5];
  const float* Wq    = (const float*)d_in[6];
  const float* Wk    = (const float*)d_in[7];
  const float* alpha = (const float*)d_in[8];
  const float* beta  = (const float*)d_in[9];
  const int*   esrc  = (const int*)d_in[10];
  const int*   edst  = (const int*)d_in[11];
  float* out = (float*)d_out;
  char* ws = (char*)d_ws;

  bf16_t*   XBF  = (bf16_t*)(ws + OFF_A);
  float*    Q    = (float*)(ws + OFF_A);     // reuse after GEMM1
  float*    LOG  = (float*)(ws + OFF_A);     // reuse after scores
  float*    SUP1 = (float*)(ws + OFF_B);
  float*    K2   = (float*)(ws + OFF_B);     // reuse after SpMM1
  float*    SUP2 = (float*)(ws + OFF_B);     // reuse after scores
  float*    H1   = (float*)(ws + OFF_C);
  bf16_t*   HBF  = (bf16_t*)(ws + OFF_D);
  float*    HCUR = (float*)(ws + OFF_E);
  float*    AGG  = (float*)(ws + OFF_F);
  float*    SC   = (float*)(ws + OFF_G);
  unsigned* MX   = (unsigned*)(ws + OFF_H);
  float*    Z    = (float*)(ws + OFF_I);
  bf16_t*   W1T  = (bf16_t*)(ws + OFF_W1T);
  bf16_t*   WQT  = (bf16_t*)(ws + OFF_WQT);
  bf16_t*   WKT  = (bf16_t*)(ws + OFF_WKT);
  bf16_t*   W2T  = (bf16_t*)(ws + OFF_W2T);
  float*    B2P  = (float*)(ws + OFF_B2P);

  const int edgeBlocks = (NEDGE + 7) / 8;        // 8 waves (edges) per 256-thr block
  const int mTiles = NNODE / 16;                 // 3125

  // --- prep: bf16 conversions + weight transposes ---
  {
    long n = (long)NNODE * NFEAT;
    k_f2bf<<<(unsigned)((n + 255) / 256), 256, 0, stream>>>(x, XBF, n);
  }
  k_wt_bf16<<<(NFEAT * NHID + 255) / 256, 256, 0, stream>>>(W1, W1T, NFEAT, NHID, NHID);
  k_wt_bf16<<<(NHID * NHID + 255) / 256, 256, 0, stream>>>(Wq, WQT, NHID, NHID, NHID);
  k_wt_bf16<<<(NHID * NHID + 255) / 256, 256, 0, stream>>>(Wk, WKT, NHID, NHID, NHID);
  k_wt_bf16<<<(NHID * NCLSP + 255) / 256, 256, 0, stream>>>(W2, W2T, NHID, NCLS, NCLSP);
  k_padb<<<1, 64, 0, stream>>>(b2, B2P, NCLS, NCLSP);

  // --- GCN layer 1: support = x@W1+b1 ; h1 = relu(scatter-add) ---
  k_wmma_gemm<<<dim3(mTiles, 1), 256, 0, stream>>>(XBF, W1T, b1, SUP1, NNODE, NFEAT, NHID);
  hipMemsetAsync(H1, 0, (size_t)NNODE * NHID * 4, stream);
  k_spmm128<<<edgeBlocks, 256, 0, stream>>>(SUP1, esrc, edst, ew, H1);
  k_relu<<<(unsigned)(((long)NNODE * NHID + 255) / 256), 256, 0, stream>>>(H1, (long)NNODE * NHID);

  // --- CRF: q = h1@Wq, k = h1@Wk, edge attention softmax ---
  {
    long n = (long)NNODE * NHID;
    k_f2bf<<<(unsigned)((n + 255) / 256), 256, 0, stream>>>(H1, HBF, n);
  }
  k_wmma_gemm<<<dim3(mTiles, 1), 256, 0, stream>>>(HBF, WQT, nullptr, Q, NNODE, NHID, NHID);
  k_wmma_gemm<<<dim3(mTiles, 1), 256, 0, stream>>>(HBF, WKT, nullptr, K2, NNODE, NHID, NHID);
  hipMemsetAsync(MX, 0, (size_t)NNODE * 4, stream);   // 0 is identity under map_f
  hipMemsetAsync(Z, 0, (size_t)NNODE * 4, stream);
  k_edge_scores<<<edgeBlocks, 256, 0, stream>>>(Q, K2, esrc, edst, SC, MX);
  k_edge_exp<<<(NEDGE + 255) / 256, 256, 0, stream>>>(SC, edst, MX, Z);

  // --- 2 mean-field iterations ---
  hipMemsetAsync(AGG, 0, (size_t)NNODE * NHID * 4, stream);
  k_crf_agg<<<edgeBlocks, 256, 0, stream>>>(H1, SC, Z, esrc, edst, AGG);
  k_crf_update<<<(unsigned)(((long)NNODE * NHID + 255) / 256), 256, 0, stream>>>(
      H1, AGG, alpha, beta, HCUR, (long)NNODE * NHID);
  hipMemsetAsync(AGG, 0, (size_t)NNODE * NHID * 4, stream);
  k_crf_agg<<<edgeBlocks, 256, 0, stream>>>(HCUR, SC, Z, esrc, edst, AGG);
  k_crf_update<<<(unsigned)(((long)NNODE * NHID + 255) / 256), 256, 0, stream>>>(
      H1, AGG, alpha, beta, HCUR, (long)NNODE * NHID);   // final h2 in HCUR

  // --- GCN layer 2 + log_softmax ---
  {
    long n = (long)NNODE * NHID;
    k_f2bf<<<(unsigned)((n + 255) / 256), 256, 0, stream>>>(HCUR, HBF, n);
  }
  k_wmma_gemm<<<dim3(mTiles, 1), 256, 0, stream>>>(HBF, W2T, B2P, SUP2, NNODE, NHID, NCLSP);
  hipMemsetAsync(LOG, 0, (size_t)NNODE * NCLS * 4, stream);
  k_spmm40<<<edgeBlocks, 256, 0, stream>>>(SUP2, esrc, edst, ew, LOG);
  k_log_softmax<<<(NNODE + 7) / 8, 256, 0, stream>>>(LOG, out);
}
